// ForWard_68375879352733
// MI455X (gfx1250) — matrix operand order repaired and checked by reading, not compiled, not run on hardware
//
#include <hip/hip_runtime.h>

// ---------------- problem constants (match reference) ----------------
constexpr int Bn  = 4096;   // batch
constexpr int Fn  = 1024;   // features (in = out)
constexpr int NLn = 24;     // hidden layers

// ---------------- CDNA5 WMMA vector types ----------------
typedef __attribute__((ext_vector_type(16))) __bf16 v16bf;
typedef __attribute__((ext_vector_type(8)))  __bf16 v8bf;
typedef __attribute__((ext_vector_type(8)))  float  v8f;

// =====================================================================
// Prep 1: Wt[l][n][k] = bf16( W[l][k][n] * M[l][k][n] )   (LDS-tiled
// transpose so both the f32 reads and the bf16 writes are coalesced).
// =====================================================================
__global__ __launch_bounds__(256) void prep_weights_kernel(
    const float* __restrict__ W, const float* __restrict__ M,
    __bf16* __restrict__ Wt) {
  __shared__ __bf16 tile[32][33];           // +1 pad: no bank conflicts
  const int l  = blockIdx.z;
  const int k0 = blockIdx.x * 32;
  const int n0 = blockIdx.y * 32;
  const int tx = threadIdx.x;               // 0..31
  const int ty = threadIdx.y;               // 0..7

#pragma unroll
  for (int j = 0; j < 4; ++j) {
    const int kl  = ty + j * 8;
    const int idx = ((l * 1025) + (k0 + kl)) * Fn + (n0 + tx);
    tile[kl][tx] = (__bf16)(W[idx] * M[idx]);
  }
  __syncthreads();
#pragma unroll
  for (int j = 0; j < 4; ++j) {
    const int nl = ty + j * 8;
    Wt[((size_t)l * Fn + (n0 + nl)) * Fn + (k0 + tx)] = tile[tx][nl];
  }
}

// Prep 2: bias[l][n] = W[l][1024][n]  (mask row is forced to 1)
__global__ __launch_bounds__(256) void prep_bias_kernel(
    const float* __restrict__ W, float* __restrict__ bias) {
  const int t = blockIdx.x * blockDim.x + threadIdx.x;
  if (t >= NLn * Fn) return;
  const int l = t >> 10, n = t & (Fn - 1);
  bias[t] = W[((size_t)l * 1025 + 1024) * Fn + n];
}

// Prep 3: act0 = bf16(x)
__global__ __launch_bounds__(256) void convert_x_kernel(
    const float* __restrict__ x, __bf16* __restrict__ a0) {
  const int t = blockIdx.x * blockDim.x + threadIdx.x;
  a0[t] = (__bf16)x[t];
}

// =====================================================================
// One layer:  out = ReLU( A(4096x1024) * Wt^T + bias )
//   A   : bf16, row-major [b][k]
//   Wt  : bf16, [n][k]  (pre-masked, pre-transposed -> contiguous K)
//   bias: f32 accumulator init
// Block = 256 thr = 8 waves (2x4), wave tile 64x64, block tile 128x256.
// K-step 32 -> 16x v_wmma_f32_16x16x32_bf16.  A sched_barrier(0) pins
// all 16 fragment loads above the MAC block so the B fragments stay
// live and the WMMA groups get progressive s_wait_loadcnt instead of
// per-group full-latency stalls.
// =====================================================================
__global__ __launch_bounds__(256) void layer_gemm_kernel(
    const __bf16* __restrict__ A, const __bf16* __restrict__ Wt,
    const float* __restrict__ bias, __bf16* __restrict__ Obf,
    float* __restrict__ Of32, int last) {
  const int lane = threadIdx.x & 31;
  const int half = lane >> 4;      // 0: lanes 0-15, 1: lanes 16-31
  const int lh   = lane & 15;
  const int wave = threadIdx.x >> 5;

  const int rowBase = blockIdx.y * 128 + (wave >> 2) * 64;  // M
  const int colBase = blockIdx.x * 256 + (wave & 3) * 64;   // N

  // Per-lane base pointers (k0 = 0), advanced by k each iteration.
  const __bf16* aPtr[4];
  const __bf16* bPtr[4];
#pragma unroll
  for (int m = 0; m < 4; ++m)
    aPtr[m] = A + (size_t)(rowBase + m * 16 + lh) * Fn + half * 16;
#pragma unroll
  for (int n = 0; n < 4; ++n)
    bPtr[n] = Wt + (size_t)(colBase + n * 16 + lh) * Fn + half * 16;

  // ---- accumulators, initialized with the bias row (all 8 rows of a
  //      16x16 C tile in a lane share the same N => same bias value) ----
  v8f acc[4][4];
#pragma unroll
  for (int n = 0; n < 4; ++n) {
    const float bv = bias[colBase + n * 16 + lh];
#pragma unroll
    for (int m = 0; m < 4; ++m) {
      v8f c;
#pragma unroll
      for (int i = 0; i < 8; ++i) c[i] = bv;
      acc[m][n] = c;
    }
  }

  // ---- K loop ----
  for (int k0 = 0; k0 < Fn; k0 += 32) {
    // A fragments (contiguous layout, matching B): lane m holds
    // K[k0+half*16 .. +15] in V0-7.
    v16bf aF[4];
#pragma unroll
    for (int m = 0; m < 4; ++m) {
      v16bf af;
      ((v8bf*)&af)[0] = *(const v8bf*)(aPtr[m] + k0);
      ((v8bf*)&af)[1] = *(const v8bf*)(aPtr[m] + k0 + 8);
      aF[m] = af;
    }
    // B fragments (contiguous layout, same as A): lane n holds 16 contiguous K of
    // column n -> single 32B load thanks to the pre-transpose.
    v16bf bF[4];
#pragma unroll
    for (int n = 0; n < 4; ++n) bF[n] = *(const v16bf*)(bPtr[n] + k0);

    // Pin every fragment load above the MACs: forbids the scheduler
    // from sinking B loads into the WMMA block (which would re-expose
    // one full load latency per 4-WMMA group).
    __builtin_amdgcn_sched_barrier(0);

#pragma unroll
    for (int n = 0; n < 4; ++n)
#pragma unroll
      for (int m = 0; m < 4; ++m)
        acc[m][n] = __builtin_amdgcn_wmma_f32_16x16x32_bf16(
            false, aF[m], false, bF[n], (short)0, acc[m][n], false, false);
  }

  // ---- ReLU + store.  C layout: VGPR r -> M = r + 8*half, N = lh ----
  if (last) {
#pragma unroll
    for (int m = 0; m < 4; ++m)
#pragma unroll
      for (int n = 0; n < 4; ++n) {
        const int col = colBase + n * 16 + lh;
#pragma unroll
        for (int r = 0; r < 8; ++r) {
          const int row = rowBase + m * 16 + half * 8 + r;
          float v = acc[m][n][r];
          Of32[(size_t)row * Fn + col] = v > 0.0f ? v : 0.0f;
        }
      }
  } else {
#pragma unroll
    for (int m = 0; m < 4; ++m)
#pragma unroll
      for (int n = 0; n < 4; ++n) {
        const int col = colBase + n * 16 + lh;
#pragma unroll
        for (int r = 0; r < 8; ++r) {
          const int row = rowBase + m * 16 + half * 8 + r;
          float v = acc[m][n][r];
          Obf[(size_t)row * Fn + col] = (__bf16)(v > 0.0f ? v : 0.0f);
        }
      }
  }
}

// =====================================================================
// Host driver.  Inputs: x, W, M, sources, layers_idx, preds_idx.
// Index arrays are identity rank-to-rank maps => unused.
// Workspace layout (needs ~67 MB):
//   Wt   : NL*F*F bf16      (50.3 MB)  -- resident in 192 MB L2
//   bias : NL*F   f32       ( 0.1 MB)
//   actA : B*F    bf16      ( 8.4 MB)
//   actB : B*F    bf16      ( 8.4 MB)
// =====================================================================
extern "C" void kernel_launch(void* const* d_in, const int* in_sizes, int n_in,
                              void* d_out, int out_size, void* d_ws,
                              size_t ws_size, hipStream_t stream) {
  const float* x = (const float*)d_in[0];
  const float* W = (const float*)d_in[1];
  const float* M = (const float*)d_in[2];
  float* out = (float*)d_out;

  char* ws = (char*)d_ws;
  size_t off = 0;
  __bf16* Wt = (__bf16*)(ws + off);
  off += (size_t)NLn * Fn * Fn * sizeof(__bf16);
  float* bias = (float*)(ws + off);
  off += (size_t)NLn * Fn * sizeof(float);
  __bf16* actA = (__bf16*)(ws + off);
  off += (size_t)Bn * Fn * sizeof(__bf16);
  __bf16* actB = (__bf16*)(ws + off);

  // prep (stream-ordered; safe for graph capture)
  prep_weights_kernel<<<dim3(Fn / 32, Fn / 32, NLn), dim3(32, 8), 0, stream>>>(
      W, M, Wt);
  prep_bias_kernel<<<(NLn * Fn + 255) / 256, 256, 0, stream>>>(W, bias);
  convert_x_kernel<<<(Bn * Fn) / 256, 256, 0, stream>>>(x, actA);

  // 24 chained masked GEMMs, ping-pong bf16 activations
  __bf16* src = actA;
  __bf16* dst = actB;
  for (int l = 0; l < NLn; ++l) {
    const int last = (l == NLn - 1);
    layer_gemm_kernel<<<dim3(Fn / 256, Bn / 128), 256, 0, stream>>>(
        src, Wt + (size_t)l * Fn * Fn, bias + (size_t)l * Fn, dst, out, last);
    __bf16* t = src; src = dst; dst = t;
  }
}
